// AttentionWithLlamaRotaryEncoding_53609781789149
// MI455X (gfx1250) — matrix operand order repaired and checked by reading, not compiled, not run on hardware
//
#include <hip/hip_runtime.h>
#include <stdint.h>

// ---------------------------------------------------------------------------
// MI455X (gfx1250) attention + Llama RoPE.
// Compute-bound (~69 GFLOP, all data resident in 192MB L2) -> bf16 WMMA
// (16x16x32, f32 accum) for every matmul; f32 softmax/RoPE math.
// Attention stages K/V chunks into LDS with global_load_async_to_lds_b128
// (ASYNCcnt, double-buffered, block-cooperative: 8x less L2 traffic).
// Workspace usage: 80 MB.
// ---------------------------------------------------------------------------

typedef __attribute__((ext_vector_type(16))) __bf16 v16bf;
typedef __attribute__((ext_vector_type(8)))  float  v8f;

#define D_MODEL 1024
#define N_HEADS 16
#define HD      64
#define SEQ     2048
#define BATCH   2

__device__ __forceinline__ unsigned short f2bf(float f) {
  unsigned int u = __float_as_uint(f);
  u += 0x7FFFu + ((u >> 16) & 1u);          // round-to-nearest-even
  return (unsigned short)(u >> 16);
}

union AB16 { v16bf v; uint4 q[2]; };

// B fragment (32x16 bf16): lane = N column, 16 consecutive K halves -> 32B.
__device__ __forceinline__ v16bf load_b_frag(const unsigned short* p) {
  AB16 f;
  f.q[0] = *(const uint4*)(const void*)(p);
  f.q[1] = *(const uint4*)(const void*)(p + 8);
  return f.v;
}
// A fragment (16x32 bf16): lane = (M, half); two disjoint 16B chunks.
__device__ __forceinline__ v16bf load_a_frag(const unsigned short* p0,
                                             const unsigned short* p1) {
  AB16 f;
  f.q[0] = *(const uint4*)(const void*)(p0);
  f.q[1] = *(const uint4*)(const void*)(p1);
  return f.v;
}

// Async global->LDS copy, 16 bytes per lane, tracked by ASYNCcnt.
#define ASYNC_B128(ldsaddr, gaddr)                                            \
  asm volatile("global_load_async_to_lds_b128 %0, %1, off" ::"v"(ldsaddr),    \
               "v"(gaddr)                                                     \
               : "memory")

// ---------------------------------------------------------------------------
__global__ void k_convert(const float* __restrict__ in,
                          unsigned short* __restrict__ out, int n) {
  int i = blockIdx.x * blockDim.x + threadIdx.x;
  if (i < n) out[i] = f2bf(in[i]);
}

// ---------------------------------------------------------------------------
// QKV GEMM: C[M=4096, N=3072] = X[4096,1024] * Wqkv^T, Wqkv row-major [N,K].
// Scatter: q,k -> f32 [B,H,S,hd] (pre-RoPE), v -> bf16 [B,H,hd,S] (transposed).
__global__ __launch_bounds__(256) void k_qkv_gemm(
    const unsigned short* __restrict__ X, const unsigned short* __restrict__ W,
    float* __restrict__ qraw, float* __restrict__ kraw,
    unsigned short* __restrict__ vtb) {
  const int lane = threadIdx.x & 31;
  const int wave = threadIdx.x >> 5;
  const int wm = wave & 3, wn = wave >> 2;
  const int r0 = blockIdx.y * 128 + wm * 32;
  const int c0 = blockIdx.x * 128 + wn * 64;
  const int n0 = lane & 15, half = lane >> 4;

  v8f acc[2][4];
#pragma unroll
  for (int i = 0; i < 2; ++i)
#pragma unroll
    for (int j = 0; j < 4; ++j) acc[i][j] = {};

  for (int kk = 0; kk < D_MODEL; kk += 32) {
    v16bf a[2], b[4];
#pragma unroll
    for (int i = 0; i < 2; ++i) {
      const unsigned short* p =
          X + (size_t)(r0 + i * 16 + n0) * D_MODEL + kk + half * 16;
      a[i] = load_a_frag(p, p + 8);
    }
#pragma unroll
    for (int j = 0; j < 4; ++j) {
      const unsigned short* p =
          W + (size_t)(c0 + j * 16 + n0) * D_MODEL + kk + half * 16;
      b[j] = load_b_frag(p);
    }
#pragma unroll
    for (int i = 0; i < 2; ++i)
#pragma unroll
      for (int j = 0; j < 4; ++j)
        acc[i][j] = __builtin_amdgcn_wmma_f32_16x16x32_bf16(
            false, a[i], false, b[j], (short)0, acc[i][j], false, false);
  }

#pragma unroll
  for (int i = 0; i < 2; ++i) {
#pragma unroll
    for (int j = 0; j < 4; ++j) {
      const int col = c0 + j * 16 + n0;
#pragma unroll
      for (int r = 0; r < 8; ++r) {
        const int row = r0 + i * 16 + r + 8 * half;   // C layout: M=r / r+8
        const int bi = row >> 11, s = row & (SEQ - 1);
        const float val = acc[i][j][r];
        if (col < D_MODEL) {
          int h = col >> 6, d = col & 63;
          qraw[((size_t)(bi * N_HEADS + h) * SEQ + s) * HD + d] = val;
        } else if (col < 2 * D_MODEL) {
          int c2 = col - D_MODEL, h = c2 >> 6, d = c2 & 63;
          kraw[((size_t)(bi * N_HEADS + h) * SEQ + s) * HD + d] = val;
        } else {
          int c2 = col - 2 * D_MODEL, h = c2 >> 6, d = c2 & 63;
          vtb[((size_t)(bi * N_HEADS + h) * HD + d) * SEQ + s] = f2bf(val);
        }
      }
    }
  }
}

// ---------------------------------------------------------------------------
// RoPE (Llama split-half variant): out[j]=x1*c-x2*s, out[j+32]=x1*s+x2*c,
// x1=in[2j], x2=in[2j+1]. scale folds in 1/sqrt(hd) for q.
__global__ void k_rope(const float* __restrict__ in,
                       unsigned short* __restrict__ out, float scale) {
  int t = blockIdx.x * blockDim.x + threadIdx.x;   // B*H*S*32 threads
  int j = t & 31;
  int s = (t >> 5) & (SEQ - 1);
  int bh = t >> 16;                                // S*32 == 2^16
  size_t base = ((size_t)bh * SEQ + s) * HD;
  float freq = __expf(-(float)j * (9.210340371976184f / 32.0f)); // 10000^(-j/32)
  float ang = (float)s * freq;
  float c = cosf(ang), sn = sinf(ang);
  float x1 = in[base + 2 * j], x2 = in[base + 2 * j + 1];
  out[base + j]      = f2bf((x1 * c - x2 * sn) * scale);
  out[base + j + 32] = f2bf((x1 * sn + x2 * c) * scale);
}

// ---------------------------------------------------------------------------
// Flash attention. One wave per 16-row q tile; all 8 waves of a block share
// one (b,h). Per 64-wide K chunk: K (8KB, [kpos][hd]) and V (8KB, [hd][kpos])
// are staged block-cooperatively into double-buffered LDS via async copies
// (4 x b128 per lane, ASYNCcnt), overlapping the next chunk's fetch with the
// current chunk's WMMA work. Online softmax in f32; P re-layout (C->A)
// through per-wave LDS.
__global__ __launch_bounds__(256) void k_attn(
    const unsigned short* __restrict__ Q, const unsigned short* __restrict__ K,
    const unsigned short* __restrict__ Vt, unsigned short* __restrict__ ctx) {
  __shared__ unsigned short kbuf[2][64 * HD];   // 2 x 8 KB, [kpos][hd]
  __shared__ unsigned short vbuf[2][64 * HD];   // 2 x 8 KB, [hd][kpos]
  __shared__ unsigned short pbuf[8][16 * HD];   // 16 KB, per-wave P tiles
  const int tid = threadIdx.x;
  const int lane = tid & 31, wave = tid >> 5;
  const int n0 = lane & 15, half = lane >> 4;
  const int bh = blockIdx.x >> 4;                  // 0..31
  const int q0 = ((blockIdx.x & 15) * 8 + wave) * 16;
  const unsigned short* qbase = Q + ((size_t)bh * SEQ + q0) * HD;
  const unsigned short* kbase = K + (size_t)bh * SEQ * HD;
  const unsigned short* vbase = Vt + (size_t)bh * HD * SEQ;
  unsigned short* lds = pbuf[wave];

  // Stage one 64-wide chunk (K + V) into LDS buffer `buf`.
  auto stage = [&](int kc, int buf) {
#pragma unroll
    for (int it = 0; it < 2; ++it) {
      const int seg = tid + it * 256;                 // 512 x 16B segments
      // K chunk is fully contiguous in global ([S][hd] row-major).
      unsigned kl = (unsigned)(size_t)&kbuf[buf][seg * 8];
      unsigned long long kg =
          (unsigned long long)(size_t)(kbase + (size_t)kc * HD + seg * 8);
      ASYNC_B128(kl, kg);
      // V chunk: 64 rows (hd) of 128B, row stride SEQ*2 bytes.
      const int row = seg >> 3, c8 = (seg & 7) * 8;
      unsigned vl = (unsigned)(size_t)&vbuf[buf][seg * 8];
      unsigned long long vg =
          (unsigned long long)(size_t)(vbase + (size_t)row * SEQ + kc + c8);
      ASYNC_B128(vl, vg);
    }
  };

  v16bf aq[2];                                     // whole q tile resident
#pragma unroll
  for (int ks = 0; ks < 2; ++ks) {
    const unsigned short* p = qbase + (size_t)n0 * HD + ks * 32 + half * 16;
    aq[ks] = load_a_frag(p, p + 8);
  }

  float mrow[8], lrow[8];
  v8f o[4];
#pragma unroll
  for (int r = 0; r < 8; ++r) { mrow[r] = -1e30f; lrow[r] = 0.f; }
#pragma unroll
  for (int t = 0; t < 4; ++t) o[t] = {};

  stage(0, 0);
  for (int ic = 0; ic < SEQ / 64; ++ic) {
    const int cur = ic & 1;
    // My async copies for chunk `ic` done; barrier publishes them to the
    // block AND guarantees everyone finished reading buffer `cur^1` from
    // chunk ic-1, so it is safe to start overwriting it with chunk ic+1.
    asm volatile("s_wait_asynccnt 0x0" ::: "memory");
    __syncthreads();
    if (ic + 1 < SEQ / 64) stage((ic + 1) * 64, cur ^ 1);

    const unsigned short* kt = kbuf[cur];
    const unsigned short* vt = vbuf[cur];

    v8f s[4];                                      // scores 16 x 64 (f32)
#pragma unroll
    for (int j = 0; j < 4; ++j) {
      v8f z = {};
      const unsigned short* kp = kt + (size_t)(j * 16 + n0) * HD + half * 16;
      v16bf b0 = load_b_frag(kp);        // hd 0..31 slice
      v16bf b1 = load_b_frag(kp + 32);   // hd 32..63 slice
      z = __builtin_amdgcn_wmma_f32_16x16x32_bf16(false, aq[0], false, b0,
                                                  (short)0, z, false, false);
      z = __builtin_amdgcn_wmma_f32_16x16x32_bf16(false, aq[1], false, b1,
                                                  (short)0, z, false, false);
      s[j] = z;
    }
    // Online softmax. C layout: lanes 0-15 hold rows r, lanes 16-31 rows r+8;
    // butterfly over xor {1,2,4,8} reduces across the 16 N-lanes of each half.
    float newm[8], scl[8];
#pragma unroll
    for (int r = 0; r < 8; ++r) {
      float v = fmaxf(fmaxf(s[0][r], s[1][r]), fmaxf(s[2][r], s[3][r]));
#pragma unroll
      for (int off = 1; off < 16; off <<= 1) v = fmaxf(v, __shfl_xor(v, off, 32));
      newm[r] = fmaxf(mrow[r], v);
      scl[r] = __expf(mrow[r] - newm[r]);
      mrow[r] = newm[r];
    }
    float lp[8];
#pragma unroll
    for (int r = 0; r < 8; ++r) lp[r] = 0.f;
#pragma unroll
    for (int j = 0; j < 4; ++j)
#pragma unroll
      for (int r = 0; r < 8; ++r) {
        float p = __expf(s[j][r] - newm[r]);
        s[j][r] = p;
        lp[r] += p;
      }
#pragma unroll
    for (int r = 0; r < 8; ++r) {
      float v = lp[r];
#pragma unroll
      for (int off = 1; off < 16; off <<= 1) v += __shfl_xor(v, off, 32);
      lrow[r] = lrow[r] * scl[r] + v;
    }
#pragma unroll
    for (int t = 0; t < 4; ++t)
#pragma unroll
      for (int r = 0; r < 8; ++r) o[t][r] *= scl[r];
    // P (C layout) -> per-wave LDS row-major [16][64] bf16.
#pragma unroll
    for (int j = 0; j < 4; ++j)
#pragma unroll
      for (int r = 0; r < 8; ++r)
        lds[(r + 8 * half) * HD + j * 16 + n0] = f2bf(s[j][r]);
    asm volatile("s_wait_dscnt 0x0" ::: "memory");  // same-wave LDS in-order
    // O += P @ V  (A frags of P from LDS; B frags from staged V chunk).
#pragma unroll
    for (int ks = 0; ks < 2; ++ks) {
      const unsigned short* pp = lds + n0 * HD + ks * 32 + half * 16;
      v16bf ap = load_a_frag(pp, pp + 8);
#pragma unroll
      for (int t = 0; t < 4; ++t) {
        const unsigned short* vp =
            vt + (size_t)(t * 16 + n0) * HD + ks * 32 + half * 16;
        v16bf bv = load_b_frag(vp);
        o[t] = __builtin_amdgcn_wmma_f32_16x16x32_bf16(false, ap, false, bv,
                                                       (short)0, o[t], false, false);
      }
    }
  }
  const int b = bh >> 4, h = bh & 15;
#pragma unroll
  for (int t = 0; t < 4; ++t) {
    const int col = h * HD + t * 16 + n0;
#pragma unroll
    for (int r = 0; r < 8; ++r) {
      const int srow = q0 + r + 8 * half;
      ctx[(size_t)(b * SEQ + srow) * D_MODEL + col] = f2bf(o[t][r] / lrow[r]);
    }
  }
}

// ---------------------------------------------------------------------------
// Output projection: out = ctx @ Wout^T + b_out  (f32 result).
__global__ __launch_bounds__(256) void k_out_gemm(
    const unsigned short* __restrict__ Xc, const unsigned short* __restrict__ W,
    const float* __restrict__ bias, float* __restrict__ out) {
  const int lane = threadIdx.x & 31, wave = threadIdx.x >> 5;
  const int wm = wave & 3, wn = wave >> 2;
  const int r0 = blockIdx.y * 128 + wm * 32;
  const int c0 = blockIdx.x * 128 + wn * 64;
  const int n0 = lane & 15, half = lane >> 4;

  v8f acc[2][4];
#pragma unroll
  for (int i = 0; i < 2; ++i)
#pragma unroll
    for (int j = 0; j < 4; ++j) acc[i][j] = {};

  for (int kk = 0; kk < D_MODEL; kk += 32) {
    v16bf a[2], b[4];
#pragma unroll
    for (int i = 0; i < 2; ++i) {
      const unsigned short* p =
          Xc + (size_t)(r0 + i * 16 + n0) * D_MODEL + kk + half * 16;
      a[i] = load_a_frag(p, p + 8);
    }
#pragma unroll
    for (int j = 0; j < 4; ++j) {
      const unsigned short* p =
          W + (size_t)(c0 + j * 16 + n0) * D_MODEL + kk + half * 16;
      b[j] = load_b_frag(p);
    }
#pragma unroll
    for (int i = 0; i < 2; ++i)
#pragma unroll
      for (int j = 0; j < 4; ++j)
        acc[i][j] = __builtin_amdgcn_wmma_f32_16x16x32_bf16(
            false, a[i], false, b[j], (short)0, acc[i][j], false, false);
  }
#pragma unroll
  for (int j = 0; j < 4; ++j) {
    const int col = c0 + j * 16 + n0;
    const float bj = bias[col];
#pragma unroll
    for (int i = 0; i < 2; ++i)
#pragma unroll
      for (int r = 0; r < 8; ++r) {
        const int row = r0 + i * 16 + r + 8 * half;
        out[(size_t)row * D_MODEL + col] = acc[i][j][r] + bj;
      }
  }
}

// ---------------------------------------------------------------------------
extern "C" void kernel_launch(void* const* d_in, const int* in_sizes, int n_in,
                              void* d_out, int out_size, void* d_ws,
                              size_t ws_size, hipStream_t stream) {
  (void)in_sizes; (void)n_in; (void)out_size; (void)ws_size;
  const float* x     = (const float*)d_in[0];
  const float* w_qkv = (const float*)d_in[1];
  const float* w_out = (const float*)d_in[2];
  const float* b_out = (const float*)d_in[3];

  char* ws = (char*)d_ws;
  const size_t MB = 1024 * 1024;
  unsigned short* xb    = (unsigned short*)(ws + 0);        //  8 MB x bf16
  unsigned short* wqkvb = (unsigned short*)(ws + 8 * MB);   //  6 MB
  unsigned short* woutb = (unsigned short*)(ws + 14 * MB);  //  2 MB
  float*          qraw  = (float*)(ws + 16 * MB);           // 16 MB pre-RoPE q
  float*          kraw  = (float*)(ws + 32 * MB);           // 16 MB pre-RoPE k
  unsigned short* qb    = (unsigned short*)(ws + 48 * MB);  //  8 MB roped q*0.125
  unsigned short* kb    = (unsigned short*)(ws + 56 * MB);  //  8 MB roped k
  unsigned short* vtb   = (unsigned short*)(ws + 64 * MB);  //  8 MB V^T [B,H,hd,S]
  unsigned short* ctxb  = (unsigned short*)(ws + 72 * MB);  //  8 MB attn output

  k_convert<<<(4096 * 1024) / 256, 256, 0, stream>>>(x, xb, 4096 * 1024);
  k_convert<<<(3072 * 1024) / 256, 256, 0, stream>>>(w_qkv, wqkvb, 3072 * 1024);
  k_convert<<<(1024 * 1024) / 256, 256, 0, stream>>>(w_out, woutb, 1024 * 1024);

  k_qkv_gemm<<<dim3(24, 32), 256, 0, stream>>>(xb, wqkvb, qraw, kraw, vtb);

  const int rope_threads = BATCH * N_HEADS * SEQ * 32;       // 2^21
  k_rope<<<rope_threads / 256, 256, 0, stream>>>(qraw, qb, 0.125f);
  k_rope<<<rope_threads / 256, 256, 0, stream>>>(kraw, kb, 1.0f);

  k_attn<<<512, 256, 0, stream>>>(qb, kb, vtb, ctxb);

  k_out_gemm<<<dim3(8, 32), 256, 0, stream>>>(ctxb, woutb, b_out, (float*)d_out);
}
